// MultiHeadAttention_63350767616242
// MI455X (gfx1250) — matrix operand order repaired and checked by reading, not compiled, not run on hardware
//
#include <hip/hip_runtime.h>
#include <hip/hip_bf16.h>

typedef _Float16 v16h __attribute__((ext_vector_type(16)));
typedef _Float16 v8h  __attribute__((ext_vector_type(8)));
typedef float    v8f  __attribute__((ext_vector_type(8)));

#define NHEADS 16
#define HEAD   64
#define TLEN   2048
#define BATCH  4
#define CDIM   1024
#define MROWS  (BATCH * TLEN)   // 8192

__device__ __forceinline__ v16h cat8(v8h lo, v8h hi) {
  return __builtin_shufflevector(lo, hi, 0, 1, 2, 3, 4, 5, 6, 7,
                                 8, 9, 10, 11, 12, 13, 14, 15);
}

__device__ __forceinline__ v8f wmma16(v16h a, v16h b, v8f c) {
  // D = A(16x32 f16) * B(32x16 f16) + C(16x16 f32)
  return __builtin_amdgcn_wmma_f32_16x16x32_f16(false, a, false, b,
                                                (short)0, c, false, false);
}

// ---------------------------------------------------------------------------
// f32 -> f16 elementwise convert
// ---------------------------------------------------------------------------
__global__ __launch_bounds__(256) void cvt_f16_k(const float* __restrict__ src,
                                                 _Float16* __restrict__ dst,
                                                 int n) {
  int i = blockIdx.x * 256 + threadIdx.x;
  if (i < n) dst[i] = (_Float16)src[i];
}

// ---------------------------------------------------------------------------
// W[k][n] (f32, CDIM x CDIM) -> Wt[n][k] (f16)  : B-operand friendly layout
// ---------------------------------------------------------------------------
__global__ __launch_bounds__(256) void transpose_f16_k(const float* __restrict__ W,
                                                       _Float16* __restrict__ Wt) {
  int i = blockIdx.x * 256 + threadIdx.x;      // over CDIM*CDIM
  int n = i & (CDIM - 1);
  int k = i >> 10;
  Wt[(size_t)n * CDIM + k] = (_Float16)W[i];
}

// ---------------------------------------------------------------------------
// Generic WMMA GEMM: C(f32, MxN) = A(f16, MxK row-major) * Bt(f16, NxK)^T + bias
// wave tile 64x64: 4 A-frags x 4 B-frags -> 16 wmma per K-step of 32.
// 16x b128 loads (8 KB) feed 262144 FLOP -> 32 FLOP per L1 byte.
// 8 waves / block cover one 64-row stripe (A loads shared through WGP$).
// ---------------------------------------------------------------------------
__global__ __launch_bounds__(256) void gemm_wmma_k(const _Float16* __restrict__ A,
                                                   const _Float16* __restrict__ Bt,
                                                   const float* __restrict__ bias,
                                                   float* __restrict__ C,
                                                   int M, int N, int K) {
  const int lane = threadIdx.x & 31;
  const int wave = (blockIdx.x << 3) + (threadIdx.x >> 5);
  const int tilesN = N >> 6;
  const int tm = wave / tilesN;
  const int tn = wave - tm * tilesN;
  if ((tm << 6) >= M) return;                  // wave-uniform
  const int m0 = tm << 6, n0 = tn << 6;
  const int r16 = lane & 15, hs = lane >> 4;

  v8f acc[4][4];
#pragma unroll
  for (int mi = 0; mi < 4; ++mi)
#pragma unroll
    for (int ni = 0; ni < 4; ++ni) acc[mi][ni] = (v8f){};

  const _Float16* ap[4];
  const _Float16* bp[4];
#pragma unroll
  for (int i = 0; i < 4; ++i) {
    ap[i] = A  + (size_t)(m0 + i * 16 + r16) * K + hs * 16;
    bp[i] = Bt + (size_t)(n0 + i * 16 + r16) * K + hs * 16;
  }

  for (int k = 0; k < K; k += 32) {
    v16h af[4], bf[4];
#pragma unroll
    for (int i = 0; i < 4; ++i)
      af[i] = cat8(*(const v8h*)(ap[i] + k), *(const v8h*)(ap[i] + k + 8));
#pragma unroll
    for (int i = 0; i < 4; ++i)
      bf[i] = cat8(*(const v8h*)(bp[i] + k), *(const v8h*)(bp[i] + k + 8));
    if (k + 288 <= K) {                        // next-slab prefetch, near cache
#pragma unroll
      for (int i = 0; i < 4; ++i) {
        __builtin_prefetch(ap[i] + k + 256, 0, 3);
        __builtin_prefetch(bp[i] + k + 256, 0, 3);
      }
    }
#pragma unroll
    for (int mi = 0; mi < 4; ++mi)
#pragma unroll
      for (int ni = 0; ni < 4; ++ni)
        acc[mi][ni] = wmma16(af[mi], bf[ni], acc[mi][ni]);
  }

#pragma unroll
  for (int ni = 0; ni < 4; ++ni) {
    const int col = n0 + ni * 16 + r16;
    const float bia = bias ? bias[col] : 0.f;
#pragma unroll
    for (int mi = 0; mi < 4; ++mi) {
#pragma unroll
      for (int r = 0; r < 8; ++r) {
        const int row = m0 + mi * 16 + r + 8 * hs;   // C layout: vgpr r -> M=r+8*half
        C[(size_t)row * N + col] = acc[mi][ni][r] + bia;
      }
    }
  }
}

// ---------------------------------------------------------------------------
// RoPE + repack:  Qf/Kf/Vf (f32, [B*T, C]) ->
//   qh,kh (f16, [B][H][T][D]) with RoPE;  vt (f16, [B][H][D][T]) transposed.
// one thread per (b,h,t,pair)
// ---------------------------------------------------------------------------
__global__ __launch_bounds__(256) void rope_pack_k(const float* __restrict__ Qf,
                                                   const float* __restrict__ Kf,
                                                   const float* __restrict__ Vf,
                                                   _Float16* __restrict__ qh,
                                                   _Float16* __restrict__ kh,
                                                   _Float16* __restrict__ vt) {
  const int idx = blockIdx.x * 256 + threadIdx.x;   // B*H*T*32 threads
  const int i = idx & 31;                           // rotary pair index
  const int t = (idx >> 5) & (TLEN - 1);
  const int h = (idx >> 16) & (NHEADS - 1);
  const int b = idx >> 20;

  const size_t row = (size_t)(b * TLEN + t) * CDIM + h * HEAD + 2 * i;
  const float inv = __powf(10000.f, -((float)(2 * i)) * (1.0f / HEAD));
  float s, c;
  __sincosf((float)t * inv, &s, &c);

  const size_t orow = ((size_t)(b * NHEADS + h) * TLEN + t) * HEAD + 2 * i;
  const float q0 = Qf[row], q1 = Qf[row + 1];
  qh[orow]     = (_Float16)(q0 * c - q1 * s);
  qh[orow + 1] = (_Float16)(q0 * s + q1 * c);
  const float k0 = Kf[row], k1 = Kf[row + 1];
  kh[orow]     = (_Float16)(k0 * c - k1 * s);
  kh[orow + 1] = (_Float16)(k0 * s + k1 * c);

  const size_t vrow = ((size_t)(b * NHEADS + h) * HEAD + 2 * i) * TLEN + t;
  vt[vrow]        = (_Float16)Vf[row];
  vt[vrow + TLEN] = (_Float16)Vf[row + 1];
}

// ---------------------------------------------------------------------------
// Flash attention: one wave = one 16-row query block of one (b,h).
// Per 32-key block: 4 score wmma + online softmax + 4 PV wmma.
// P-tile is re-laid-out C->A through per-wave LDS (s_wait_dscnt for ordering).
// Output written as f16 [B*T, C] (A-operand for the final projection GEMM).
// ---------------------------------------------------------------------------
__global__ __launch_bounds__(256) void attn_wmma_k(const _Float16* __restrict__ qh,
                                                   const _Float16* __restrict__ kh,
                                                   const _Float16* __restrict__ vt,
                                                   _Float16* __restrict__ ah) {
  __shared__ _Float16 plds[8][16 * 32];             // 8 KB, per-wave P tiles
  const int lane = threadIdx.x & 31;
  const int wid  = threadIdx.x >> 5;
  const int task = (blockIdx.x << 3) + wid;         // 8192 wave tasks
  const int qb = task & 127;
  const int bh = task >> 7;
  const int b  = bh >> 4;
  const int h  = bh & 15;
  const int m0 = qb << 4;
  const int n16 = lane & 15, hs = lane >> 4;

  const _Float16* Qb = qh + (size_t)bh * TLEN * HEAD;
  const _Float16* Kb = kh + (size_t)bh * TLEN * HEAD;
  const _Float16* Vb = vt + (size_t)bh * HEAD * TLEN;

  // Q fragments (A layout), d = 0..31 and 32..63, resident all loop.
  v16h qa0, qa1;
  {
    const _Float16* qp = Qb + (size_t)(m0 + n16) * HEAD + hs * 16;
    qa0 = cat8(*(const v8h*)qp, *(const v8h*)(qp + 8));
    qa1 = cat8(*(const v8h*)(qp + 32), *(const v8h*)(qp + 40));
  }

  v8f o0 = {}, o1 = {}, o2 = {}, o3 = {};
  float mrow[8], lrow[8];
#pragma unroll
  for (int r = 0; r < 8; ++r) { mrow[r] = -1e30f; lrow[r] = 0.f; }

  const int jmax = (m0 + 15) >> 5;                  // causal: keys <= row
  for (int j = 0; j <= jmax; ++j) {
    const int kb = j << 5;
    v8f s0 = {}, s1 = {};
    {
      // K fragments: Bt-form is kh itself ([t][d] row-major), n = key index.
      const _Float16* kp0 = Kb + (size_t)(kb + n16) * HEAD + hs * 16;
      const _Float16* kp1 = kp0 + 16 * HEAD;
      v16h b00 = cat8(*(const v8h*)kp0,        *(const v8h*)(kp0 + 8));
      v16h b01 = cat8(*(const v8h*)(kp0 + 32), *(const v8h*)(kp0 + 40));
      v16h b10 = cat8(*(const v8h*)kp1,        *(const v8h*)(kp1 + 8));
      v16h b11 = cat8(*(const v8h*)(kp1 + 32), *(const v8h*)(kp1 + 40));
      s0 = wmma16(qa0, b00, s0);
      s0 = wmma16(qa1, b01, s0);
      s1 = wmma16(qa0, b10, s1);
      s1 = wmma16(qa1, b11, s1);
    }

    // online softmax in C layout (row = r + 8*hs, col = n16 / n16+16)
#pragma unroll
    for (int r = 0; r < 8; ++r) {
      const int row = m0 + r + 8 * hs;
      float v0 = s0[r] * 0.125f;                    // D^-0.5
      float v1 = s1[r] * 0.125f;
      if (kb + n16 > row)      v0 = -1e30f;
      if (kb + 16 + n16 > row) v1 = -1e30f;
      float mx = fmaxf(v0, v1);
#pragma unroll
      for (int o = 8; o >= 1; o >>= 1) mx = fmaxf(mx, __shfl_xor(mx, o, 16));
      const float mnew = fmaxf(mrow[r], mx);
      const float corr = __expf(mrow[r] - mnew);
      const float p0 = __expf(v0 - mnew);
      const float p1 = __expf(v1 - mnew);
      float ps = p0 + p1;
#pragma unroll
      for (int o = 8; o >= 1; o >>= 1) ps += __shfl_xor(ps, o, 16);
      lrow[r] = lrow[r] * corr + ps;
      mrow[r] = mnew;
      o0[r] *= corr; o1[r] *= corr; o2[r] *= corr; o3[r] *= corr;
      plds[wid][(r + 8 * hs) * 32 + n16]      = (_Float16)p0;
      plds[wid][(r + 8 * hs) * 32 + 16 + n16] = (_Float16)p1;
    }
    __asm__ volatile("s_wait_dscnt 0" ::: "memory");

    // re-load P as A fragment (16x32) from LDS
    v16h pa;
    {
      const _Float16* pp = &plds[wid][n16 * 32 + hs * 16];
      pa = cat8(*(const v8h*)pp, *(const v8h*)(pp + 8));
    }

    // V fragments from transposed vt ([d][t]), n = d, k = key index
    {
      const _Float16* vp = Vb + (size_t)n16 * TLEN + kb + hs * 16;
      v16h vb0 = cat8(*(const v8h*)vp,                 *(const v8h*)(vp + 8));
      v16h vb1 = cat8(*(const v8h*)(vp + 16 * TLEN),   *(const v8h*)(vp + 16 * TLEN + 8));
      v16h vb2 = cat8(*(const v8h*)(vp + 32 * TLEN),   *(const v8h*)(vp + 32 * TLEN + 8));
      v16h vb3 = cat8(*(const v8h*)(vp + 48 * TLEN),   *(const v8h*)(vp + 48 * TLEN + 8));
      o0 = wmma16(pa, vb0, o0);
      o1 = wmma16(pa, vb1, o1);
      o2 = wmma16(pa, vb2, o2);
      o3 = wmma16(pa, vb3, o3);
    }
  }

  // epilogue: normalize, store f16 in [B*T, C] layout for the Wo GEMM
  _Float16* outp = ah + (size_t)(b * TLEN) * CDIM + (size_t)h * HEAD;
#pragma unroll
  for (int r = 0; r < 8; ++r) {
    const int t = m0 + r + 8 * hs;
    const float inv = 1.0f / lrow[r];
    _Float16* orow = outp + (size_t)t * CDIM;
    orow[n16]      = (_Float16)(o0[r] * inv);
    orow[16 + n16] = (_Float16)(o1[r] * inv);
    orow[32 + n16] = (_Float16)(o2[r] * inv);
    orow[48 + n16] = (_Float16)(o3[r] * inv);
  }
}

// ---------------------------------------------------------------------------
extern "C" void kernel_launch(void* const* d_in, const int* in_sizes, int n_in,
                              void* d_out, int out_size, void* d_ws, size_t ws_size,
                              hipStream_t stream) {
  const float* x  = (const float*)d_in[0];
  const float* Wq = (const float*)d_in[1];
  const float* Wk = (const float*)d_in[2];
  const float* Wv = (const float*)d_in[3];
  const float* Wo = (const float*)d_in[4];
  const float* bo = (const float*)d_in[5];
  float* out = (float*)d_out;

  char* ws = (char*)d_ws;
  const size_t MK = (size_t)MROWS * CDIM;           // 8,388,608 elems
  const size_t WW = (size_t)CDIM * CDIM;            // 1,048,576 elems

  _Float16* xh  = (_Float16*)ws; ws += MK * 2;
  _Float16* wqt = (_Float16*)ws; ws += WW * 2;
  _Float16* wkt = (_Float16*)ws; ws += WW * 2;
  _Float16* wvt = (_Float16*)ws; ws += WW * 2;
  _Float16* wot = (_Float16*)ws; ws += WW * 2;
  float* qf = (float*)ws; ws += MK * 4;
  float* kf = (float*)ws; ws += MK * 4;
  float* vf = (float*)ws; ws += MK * 4;
  _Float16* qhp = (_Float16*)ws; ws += MK * 2;
  _Float16* khp = (_Float16*)ws; ws += MK * 2;
  _Float16* vtp = (_Float16*)ws; ws += MK * 2;
  _Float16* ahp = (_Float16*)ws; ws += MK * 2;

  // 1) precision conversion / operand layout
  cvt_f16_k<<<(int)(MK / 256), 256, 0, stream>>>(x, xh, (int)MK);
  transpose_f16_k<<<(int)(WW / 256), 256, 0, stream>>>(Wq, wqt);
  transpose_f16_k<<<(int)(WW / 256), 256, 0, stream>>>(Wk, wkt);
  transpose_f16_k<<<(int)(WW / 256), 256, 0, stream>>>(Wv, wvt);
  transpose_f16_k<<<(int)(WW / 256), 256, 0, stream>>>(Wo, wot);

  // 2) QKV projections (8192x1024x1024 each), 64x64 wave tiles
  const int gemmBlocks = (MROWS / 64) * (CDIM / 64) / 8;  // 256
  gemm_wmma_k<<<gemmBlocks, 256, 0, stream>>>(xh, wqt, nullptr, qf, MROWS, CDIM, CDIM);
  gemm_wmma_k<<<gemmBlocks, 256, 0, stream>>>(xh, wkt, nullptr, kf, MROWS, CDIM, CDIM);
  gemm_wmma_k<<<gemmBlocks, 256, 0, stream>>>(xh, wvt, nullptr, vf, MROWS, CDIM, CDIM);

  // 3) RoPE + head-major repack (V transposed for PV wmma)
  const int ropeThreads = BATCH * NHEADS * TLEN * 32;      // 4,194,304
  rope_pack_k<<<ropeThreads / 256, 256, 0, stream>>>(qf, kf, vf, qhp, khp, vtp);

  // 4) causal flash attention (8192 wave tasks / 8 waves per block)
  attn_wmma_k<<<1024, 256, 0, stream>>>(qhp, khp, vtp, ahp);

  // 5) output projection + bias -> d_out (f32 [B,T,C])
  gemm_wmma_k<<<gemmBlocks, 256, 0, stream>>>(ahp, wot, bo, out, MROWS, CDIM, CDIM);
}
